// KeyFrameAttention_41927470743887
// MI455X (gfx1250) — matrix operand order repaired and checked by reading, not compiled, not run on hardware
//
#include <hip/hip_runtime.h>
#include <hip/hip_bf16.h>
#include <math.h>

// ---------------------------------------------------------------------------
// Types for WMMA (gfx1250, wave32)
// ---------------------------------------------------------------------------
typedef __attribute__((ext_vector_type(16))) __bf16 v16bf;
typedef __attribute__((ext_vector_type(8)))  __bf16 v8bf;
typedef __attribute__((ext_vector_type(8)))  float  v8f;

#define WMMA_BF16(a, b, c) \
  __builtin_amdgcn_wmma_f32_16x16x32_bf16(false, (a), false, (b), (short)0, (c), false, false)

// A-fragment: two contiguous 16-byte chunks (K at +0 and +8 for this lane half)
static __device__ __forceinline__ v16bf join16(v8bf lo, v8bf hi) {
  v16bf r;
#pragma unroll
  for (int i = 0; i < 8; ++i) { r[i] = lo[i]; r[8 + i] = hi[i]; }
  return r;
}

// ---------------------------------------------------------------------------
// f32 -> bf16 (optionally scaled)
// ---------------------------------------------------------------------------
__global__ void cvt_bf16_kernel(const float* __restrict__ s, __bf16* __restrict__ d,
                                long n, float scale) {
  long i = (long)blockIdx.x * blockDim.x + threadIdx.x;
  if (i < n) d[i] = (__bf16)(s[i] * scale);
}

// f32 [R x Cc] row-major -> bf16 TRANSPOSED [Cc x R] (for GEMM B operands)
__global__ void cvt_t_bf16_kernel(const float* __restrict__ s, __bf16* __restrict__ d,
                                  int R, int Cc, float scale) {
  long i = (long)blockIdx.x * blockDim.x + threadIdx.x;
  if (i < (long)R * Cc) {
    int r = (int)(i / Cc), c = (int)(i % Cc);
    d[(long)c * R + r] = (__bf16)(s[i] * scale);
  }
}

// ---------------------------------------------------------------------------
// Batched GEMM with transposed-B operand:
//   out[M,Nn] = A[M,K] @ Bt[Nn,K]^T (+ bias*bscale)
// All fragment loads are contiguous 32B per lane -> pure global_load_b128.
// Block: 256 thr = 8 waves; wave computes 32 rows x 64 cols (8 WMMA / K-step,
// B fragments shared across the two row tiles). K-loop unrolled x2 for load
// clustering. grid = (Nn/64, M/256, batch).
// Output: f32 (out_f), bf16 (out_bf), or bf16 transposed per-512-row group
// (out_t, used to produce v^T directly).
// ---------------------------------------------------------------------------
__global__ __launch_bounds__(256)
void gemm_bt_wmma(const __bf16* __restrict__ A, const __bf16* __restrict__ Bt,
                  const float* __restrict__ bias, float bias_scale,
                  __bf16* __restrict__ out_bf, __bf16* __restrict__ out_t,
                  float* __restrict__ out_f,
                  int Nn, int K, long strideA, long strideB, long strideO) {
  const int tid  = threadIdx.x;
  const int wave = tid >> 5;
  const int lane = tid & 31;
  const int nl   = lane & 15;
  const int n0   = blockIdx.x * 64;
  const int m0   = blockIdx.y * 256 + wave * 32;
  const long bat = blockIdx.z;

  const __bf16* Ab = A  + bat * strideA;
  const __bf16* Bb = Bt + bat * strideB;

  const int kbase = (lane >> 4) * 16;  // A-frag: 16 consecutive K values (+0 / +8)
  const int kh    = (lane >> 4) * 16;  // B-frag: 16 consecutive K values

  v8f acc[2][4] = {};
  const __bf16* arow0 = Ab + (long)(m0 + nl) * K;
  const __bf16* arow1 = Ab + (long)(m0 + 16 + nl) * K;

#pragma unroll 2
  for (int kt = 0; kt < K; kt += 32) {
    v16bf a0 = join16(*(const v8bf*)(arow0 + kt + kbase),
                      *(const v8bf*)(arow0 + kt + kbase + 8));
    v16bf a1 = join16(*(const v8bf*)(arow1 + kt + kbase),
                      *(const v8bf*)(arow1 + kt + kbase + 8));
#pragma unroll
    for (int j = 0; j < 4; ++j) {
      const __bf16* brow = Bb + (long)(n0 + j * 16 + nl) * K + kt + kh;
      v16bf bf = *(const v16bf*)brow;              // 32B contiguous, aligned
      __builtin_prefetch(brow + 64, 0, 1);         // speculative; OOB is dropped
      acc[0][j] = WMMA_BF16(a0, bf, acc[0][j]);
      acc[1][j] = WMMA_BF16(a1, bf, acc[1][j]);
    }
  }

#pragma unroll
  for (int mi = 0; mi < 2; ++mi) {
    const int rbase = m0 + mi * 16 + ((lane >> 4) << 3);
#pragma unroll
    for (int j = 0; j < 4; ++j) {
      const int col = n0 + j * 16 + nl;
      const float bv = bias ? bias[col] * bias_scale : 0.0f;
#pragma unroll
      for (int r = 0; r < 8; ++r) {
        const float val = acc[mi][j][r] + bv;
        const int row = rbase + r;
        if (out_f) {
          out_f[bat * strideO + (long)row * Nn + col] = val;
        } else if (out_t) {
          // v^T layout: [row>>9][col][row&511]  (row groups of 512)
          out_t[((long)(row >> 9) * Nn + col) * 512 + (row & 511)] = (__bf16)val;
        } else {
          out_bf[bat * strideO + (long)row * Nn + col] = (__bf16)val;
        }
      }
    }
  }
}

// ---------------------------------------------------------------------------
// Masked scores: S[b,n,m] = q[b,n,:] . k[b,m,:] (q pre-scaled by 1/sqrt(C)),
// -1e20 where Mask==0. Same fragment pattern (k rows contiguous along C).
// grid = (N/64, N/256, B), block 256.
// ---------------------------------------------------------------------------
__global__ __launch_bounds__(256)
void scores_wmma(const __bf16* __restrict__ q, const __bf16* __restrict__ k,
                 const int* __restrict__ Mask, float* __restrict__ S,
                 int Nn, int C) {
  const int tid = threadIdx.x, wave = tid >> 5, lane = tid & 31, nl = lane & 15;
  const int n0 = blockIdx.x * 64;              // S column (k-row) base
  const int m0 = blockIdx.y * 256 + wave * 32; // S row (q-row) base
  const long b = blockIdx.z;
  const __bf16* qb = q + b * (long)Nn * C;
  const __bf16* kb = k + b * (long)Nn * C;
  const int kbase = (lane >> 4) * 16;
  const int kh    = (lane >> 4) * 16;

  v8f acc[2][4] = {};
  const __bf16* arow0 = qb + (long)(m0 + nl) * C;
  const __bf16* arow1 = qb + (long)(m0 + 16 + nl) * C;

#pragma unroll 2
  for (int c0 = 0; c0 < C; c0 += 32) {
    v16bf a0 = join16(*(const v8bf*)(arow0 + c0 + kbase),
                      *(const v8bf*)(arow0 + c0 + kbase + 8));
    v16bf a1 = join16(*(const v8bf*)(arow1 + c0 + kbase),
                      *(const v8bf*)(arow1 + c0 + kbase + 8));
#pragma unroll
    for (int j = 0; j < 4; ++j) {
      const __bf16* brow = kb + (long)(n0 + j * 16 + nl) * C + c0 + kh;
      v16bf bf = *(const v16bf*)brow;
      __builtin_prefetch(brow + 64, 0, 1);
      acc[0][j] = WMMA_BF16(a0, bf, acc[0][j]);
      acc[1][j] = WMMA_BF16(a1, bf, acc[1][j]);
    }
  }

#pragma unroll
  for (int mi = 0; mi < 2; ++mi) {
    const int rbase = m0 + mi * 16 + ((lane >> 4) << 3);
#pragma unroll
    for (int j = 0; j < 4; ++j) {
      const int col = n0 + j * 16 + nl;
#pragma unroll
      for (int r = 0; r < 8; ++r) {
        long idx = (b * Nn + (rbase + r)) * (long)Nn + col;
        S[idx] = Mask[idx] ? acc[mi][j][r] : -1e20f;
      }
    }
  }
}

// ---------------------------------------------------------------------------
// Row softmax over S[b,i,:]; writes TRANSPOSED attn (At[b,j,i]) in bf16 so the
// attention-apply (out = A^T @ V) is a plain row-major GEMM.
// grid = B*N blocks, 128 threads, N == 512.
// ---------------------------------------------------------------------------
__global__ __launch_bounds__(128)
void softmax_rows(const float* __restrict__ S, __bf16* __restrict__ At, int Nn) {
  __shared__ float red[128];
  const int tid = threadIdx.x;
  const long row = blockIdx.x;         // b*Nn + i
  const long b   = row / Nn;
  const int  i   = (int)(row % Nn);
  const float* srow = S + row * (long)Nn;

  float v4[4];
  float m = -3.4e38f;
#pragma unroll
  for (int t = 0; t < 4; ++t) { v4[t] = srow[tid + t * 128]; m = fmaxf(m, v4[t]); }
  red[tid] = m; __syncthreads();
  for (int s = 64; s > 0; s >>= 1) {
    if (tid < s) red[tid] = fmaxf(red[tid], red[tid + s]);
    __syncthreads();
  }
  m = red[0]; __syncthreads();

  float sum = 0.f;
#pragma unroll
  for (int t = 0; t < 4; ++t) { v4[t] = __expf(v4[t] - m); sum += v4[t]; }
  red[tid] = sum; __syncthreads();
  for (int s = 64; s > 0; s >>= 1) {
    if (tid < s) red[tid] += red[tid + s];
    __syncthreads();
  }
  const float inv = 1.0f / red[0];
#pragma unroll
  for (int t = 0; t < 4; ++t) {
    int j = tid + t * 128;
    At[(b * Nn + j) * (long)Nn + i] = (__bf16)(v4[t] * inv);
  }
}

// ---------------------------------------------------------------------------
// Launch
// ---------------------------------------------------------------------------
extern "C" void kernel_launch(void* const* d_in, const int* in_sizes, int n_in,
                              void* d_out, int out_size, void* d_ws, size_t ws_size,
                              hipStream_t stream) {
  (void)in_sizes; (void)n_in; (void)out_size; (void)ws_size;
  constexpr int B = 64, N = 512, C = 1280;
  constexpr long BNC = (long)B * N * C;
  constexpr long BNN = (long)B * N * N;
  const float qscale = 1.0f / sqrtf((float)C);

  const float* x    = (const float*)d_in[0];
  const int*   Mask = (const int*)  d_in[1];
  const float* Wq   = (const float*)d_in[2];
  const float* bq   = (const float*)d_in[3];
  const float* Wk   = (const float*)d_in[4];
  const float* bk   = (const float*)d_in[5];
  const float* Wv   = (const float*)d_in[6];
  const float* bv   = (const float*)d_in[7];
  const float* Wr   = (const float*)d_in[8];
  const float* br   = (const float*)d_in[9];
  float* out = (float*)d_out;

  // workspace carve-up (256B aligned)
  char* ws = (char*)d_ws;
  size_t off = 0;
  auto carve = [&](size_t bytes) -> char* {
    char* p = ws + off;
    off = (off + bytes + 255) & ~(size_t)255;
    return p;
  };
  __bf16* xb   = (__bf16*)carve(BNC * 2);
  __bf16* wqt  = (__bf16*)carve((size_t)C * C * 2);   // W^T bf16
  __bf16* wkt  = (__bf16*)carve((size_t)C * C * 2);
  __bf16* wvt  = (__bf16*)carve((size_t)C * C * 2);
  __bf16* wrt  = (__bf16*)carve((size_t)C * C * 2);
  __bf16* qb   = (__bf16*)carve(BNC * 2);
  __bf16* kb   = (__bf16*)carve(BNC * 2);
  __bf16* vt   = (__bf16*)carve(BNC * 2);             // v transposed [b][c][m]
  float*  Sf   = (float*) carve(BNN * 4);
  __bf16* At   = (__bf16*)carve(BNN * 2);             // attn transposed [b][n][m]
  __bf16* attb = (__bf16*)carve(BNC * 2);

  // 1) conversions: x plain; weights transposed (1/sqrt(C) folded into Wq)
  {
    long n = BNC;
    cvt_bf16_kernel<<<dim3((unsigned)((n + 255) / 256)), dim3(256), 0, stream>>>(x, xb, n, 1.0f);
    long nw = (long)C * C;
    unsigned gw = (unsigned)((nw + 255) / 256);
    cvt_t_bf16_kernel<<<dim3(gw), dim3(256), 0, stream>>>(Wq, wqt, C, C, qscale);
    cvt_t_bf16_kernel<<<dim3(gw), dim3(256), 0, stream>>>(Wk, wkt, C, C, 1.0f);
    cvt_t_bf16_kernel<<<dim3(gw), dim3(256), 0, stream>>>(Wv, wvt, C, C, 1.0f);
    cvt_t_bf16_kernel<<<dim3(gw), dim3(256), 0, stream>>>(Wr, wrt, C, C, 1.0f);
  }

  // 2) q,k,v projections: [B*N, C] @ W (+bias); v stored transposed
  {
    dim3 grid(C / 64, (B * N) / 256, 1), blk(256);
    gemm_bt_wmma<<<grid, blk, 0, stream>>>(xb, wqt, bq, qscale,
                                           qb, nullptr, nullptr, C, C, 0, 0, 0);
    gemm_bt_wmma<<<grid, blk, 0, stream>>>(xb, wkt, bk, 1.0f,
                                           kb, nullptr, nullptr, C, C, 0, 0, 0);
    gemm_bt_wmma<<<grid, blk, 0, stream>>>(xb, wvt, bv, 1.0f,
                                           nullptr, vt, nullptr, C, C, 0, 0, 0);
  }

  // 3) masked scores: S[b] = q[b] @ k[b]^T
  {
    dim3 grid(N / 64, N / 256, B), blk(256);
    scores_wmma<<<grid, blk, 0, stream>>>(qb, kb, Mask, Sf, N, C);
  }

  // 4) softmax rows -> transposed bf16 attn
  softmax_rows<<<dim3(B * N), dim3(128), 0, stream>>>(Sf, At, N);

  // 5) att_feat[b] = At[b] (N x N) @ vt[b]^T  (vt rows are channels)
  {
    dim3 grid(C / 64, N / 256, B), blk(256);
    gemm_bt_wmma<<<grid, blk, 0, stream>>>(At, vt, nullptr, 0.0f,
                                           attb, nullptr, nullptr,
                                           C, N, (long)N * N, (long)C * N, (long)N * C);
  }

  // 6) output projection -> f32 d_out
  {
    dim3 grid(C / 64, (B * N) / 256, 1), blk(256);
    gemm_bt_wmma<<<grid, blk, 0, stream>>>(attb, wrt, br, 1.0f,
                                           nullptr, nullptr, out, C, C, 0, 0, 0);
  }
}
